// MMDLoss_7645041787451
// MI455X (gfx1250) — matrix operand order repaired and checked by reading, not compiled, not run on hardware
//
#include <hip/hip_runtime.h>

typedef __attribute__((ext_vector_type(16))) _Float16 v16h;
typedef __attribute__((ext_vector_type(8)))  _Float16 v8h;
typedef __attribute__((ext_vector_type(8)))  float    v8f;

#define NROWS 8192
#define DIM   64
#define LOG2E 1.4426950408889634f

// ws layout:
//   float  [0 .. 16383]   : row norms pre-scaled by -0.5*log2(e)  (f0 then f1)
//   float  [16384..16386] : pair accumulators (xx, yy, xy)
//   _Float16 starting at float index 16448 (byte 65792, 256B aligned):
//     hi0[8192*64], lo0[8192*64], hi1[8192*64], lo1[8192*64]   (4 MB total)
#define WS_ACC   16384
#define WS_HALF  16448
#define MAT_ELEMS (NROWS * DIM)

__global__ void mmd_prep(const float* __restrict__ f0, const float* __restrict__ f1,
                         float* __restrict__ ws) {
  int tid = blockIdx.x * blockDim.x + threadIdx.x;
  if (tid < 3) ws[WS_ACC + tid] = 0.0f;            // zero accumulators
  if (tid >= 2 * NROWS) return;
  const int m   = (tid < NROWS) ? 0 : 1;
  const int row = tid - m * NROWS;
  const float* src = m ? f1 : f0;
  _Float16* hp = (_Float16*)(ws + WS_HALF) + (size_t)m * (2 * MAT_ELEMS);
  _Float16* lp = hp + MAT_ELEMS;

  const float4* p = reinterpret_cast<const float4*>(src + (size_t)row * DIM);
  _Float16* hrow = hp + (size_t)row * DIM;
  _Float16* lrow = lp + (size_t)row * DIM;
  float s = 0.0f;
#pragma unroll
  for (int i = 0; i < DIM / 4; ++i) {
    float4 v = p[i];
    s += v.x * v.x + v.y * v.y + v.z * v.z + v.w * v.w;
    float xs[4] = {v.x, v.y, v.z, v.w};
#pragma unroll
    for (int e = 0; e < 4; ++e) {
      _Float16 h = (_Float16)xs[e];
      hrow[i * 4 + e] = h;
      lrow[i * 4 + e] = (_Float16)(xs[e] - (float)h);
    }
  }
  // Pre-fold the exp scaling: exp(-sq/2) = exp2(-0.5*log2e*(na+nb) + log2e*dot)
  ws[tid] = -0.5f * LOG2E * s;
}

// 3 pairs x 512 row-tiles x 8 column chunks = 12288 waves; each wave walks 64
// 16x16 tiles. 6 WMMAs per tile (hi/lo f16 split => fp32-class accuracy, two
// K=32 steps); fragments loaded directly as pre-split f16 from L2-resident ws.
__global__ void __launch_bounds__(256) mmd_gemm(float* __restrict__ ws) {
  const int lane = threadIdx.x & 31;
  const int wave = blockIdx.x * (blockDim.x >> 5) + (threadIdx.x >> 5);
  const int p     = wave >> 12;      // 4096 waves per pair: 0=xx, 1=yy, 2=xy
  const int rem   = wave & 4095;
  const int rt    = rem >> 3;        // row tile 0..511
  const int chunk = rem & 7;         // column chunk 0..7 (64 tiles each)
  const int row0  = rt << 4;

  const int ma = (p == 1) ? 1 : 0;   // A source matrix
  const int mb = (p == 0) ? 0 : 1;   // B source matrix
  const _Float16* hbase = (const _Float16*)(ws + WS_HALF);
  const _Float16* hiA = hbase + (size_t)ma * (2 * MAT_ELEMS);
  const _Float16* loA = hiA + MAT_ELEMS;
  const _Float16* hiB = hbase + (size_t)mb * (2 * MAT_ELEMS);
  const _Float16* loB = hiB + MAT_ELEMS;
  const float* nA = ws + ma * NROWS;
  const float* nB = ws + mb * NROWS;

  // ---- A fragments (A layout: lane holds row lane&15; K in [kb,kb+16)
  // contiguous with kb = 16*(lane>>4)); reused across all 64 column tiles ----
  const int arow = row0 + (lane & 15);
  const int kb   = (lane >> 4) << 4;             // 0 or 16
  const _Float16* ahp = hiA + (size_t)arow * DIM;
  const _Float16* alp = loA + (size_t)arow * DIM;
  v16h a_hi[2], a_lo[2];
#pragma unroll
  for (int s = 0; s < 2; ++s) {
    v8h h0 = *(const v8h*)(ahp + s * 32 + kb);
    v8h h1 = *(const v8h*)(ahp + s * 32 + kb + 8);
    v8h l0 = *(const v8h*)(alp + s * 32 + kb);
    v8h l1 = *(const v8h*)(alp + s * 32 + kb + 8);
#pragma unroll
    for (int j = 0; j < 8; ++j) {
      a_hi[s][j] = h0[j];  a_hi[s][8 + j] = h1[j];
      a_lo[s][j] = l0[j];  a_lo[s][8 + j] = l1[j];
    }
  }

  // Pre-scaled row norms for this wave's 8 output rows per lane
  // (C layout: M = v + 8*(lane>>4), N = lane&15)
  float na_v[8];
#pragma unroll
  for (int v = 0; v < 8; ++v) na_v[v] = nA[row0 + v + ((lane >> 4) << 3)];

  // B layout: lane holds column lane&15; K = khB + 0..15 contiguous, khB = 16*(lane>>4)
  const int khB = (lane >> 4) << 4;              // 0 or 16
  float local = 0.0f;

  for (int t = 0; t < 64; ++t) {
    const int col0 = ((chunk << 6) + t) << 4;
    const size_t boff = (size_t)(col0 + (lane & 15)) * DIM + khB;
    const _Float16* bhp = hiB + boff;
    const _Float16* blp = loB + boff;

    v8f c = {};
#pragma unroll
    for (int s = 0; s < 2; ++s) {
      v16h b_hi = *(const v16h*)(bhp + s * 32);  // 32B aligned, contiguous K
      v16h b_lo = *(const v16h*)(blp + s * 32);
      c = __builtin_amdgcn_wmma_f32_16x16x32_f16(false, a_hi[s], false, b_hi,
                                                 (short)0, c, false, false);
      c = __builtin_amdgcn_wmma_f32_16x16x32_f16(false, a_hi[s], false, b_lo,
                                                 (short)0, c, false, false);
      c = __builtin_amdgcn_wmma_f32_16x16x32_f16(false, a_lo[s], false, b_hi,
                                                 (short)0, c, false, false);
    }

    const float nb_n = nB[col0 + (lane & 15)];
#pragma unroll
    for (int v = 0; v < 8; ++v) {
      // norms already carry -0.5*log2e; dot gets +log2e -> one fma + raw v_exp_f32
      local += __builtin_amdgcn_exp2f(__builtin_fmaf(LOG2E, c[v], na_v[v] + nb_n));
    }
  }

  // Wave32 reduction, then one atomic per wave
#pragma unroll
  for (int off = 16; off > 0; off >>= 1)
    local += __shfl_xor(local, off, 32);
  if (lane == 0) atomicAdd(&ws[WS_ACC + p], local);
}

__global__ void mmd_finalize(const float* __restrict__ ws, float* __restrict__ out) {
  if (threadIdx.x == 0 && blockIdx.x == 0) {
    const float inv = 1.0f / (8192.0f * 8192.0f);
    out[0] = (ws[WS_ACC + 0] + ws[WS_ACC + 1] - 2.0f * ws[WS_ACC + 2]) * inv;
  }
}

extern "C" void kernel_launch(void* const* d_in, const int* in_sizes, int n_in,
                              void* d_out, int out_size, void* d_ws, size_t ws_size,
                              hipStream_t stream) {
  const float* f0 = (const float*)d_in[0];
  const float* f1 = (const float*)d_in[1];
  float* ws  = (float*)d_ws;
  float* out = (float*)d_out;

  mmd_prep<<<(2 * NROWS + 255) / 256, 256, 0, stream>>>(f0, f1, ws);
  // 12288 waves total = 1536 blocks x 8 waves
  mmd_gemm<<<1536, 256, 0, stream>>>(ws);
  mmd_finalize<<<1, 1, 0, stream>>>(ws, out);
}
